// DualPathLF_45311904973262
// MI455X (gfx1250) — matrix operand order repaired and checked by reading, not compiled, not run on hardware
//
#include <hip/hip_runtime.h>

#define B_  8
#define D_  64
#define T_  2000
#define NF_ 32
#define EPS_ 1e-5f

typedef __attribute__((ext_vector_type(16))) _Float16 v16h;
typedef __attribute__((ext_vector_type(8)))  float    v8f;
typedef _Float16 h16;

static __device__ __forceinline__ v8f vzero() {
  v8f c = {0.f,0.f,0.f,0.f,0.f,0.f,0.f,0.f};
  return c;
}

static __device__ __forceinline__ v8f wmma32(v16h a, v16h b, v8f c) {
  return __builtin_amdgcn_wmma_f32_16x16x32_f16(false, a, false, b, (short)0, c, false, false);
}

// A-fragment (16x32 f16): lane holds row m0+(lane&15); elements 0..15 -> K = 16*(lane>=16)+i
// (contiguous K run of 16).
static __device__ __forceinline__ v16h load_a_frag(const h16* A, int lda, int m0, int k0, int lane) {
  const h16* p = A + (size_t)(m0 + (lane & 15)) * lda + k0 + ((lane >> 4) << 4);
  v16h r;
#pragma unroll
  for (int i = 0; i < 8; ++i) { r[i] = p[i]; r[8 + i] = p[8 + i]; }
  return r;
}

// B-fragment (32x16 f16) from W^T stored row-major [N][K]: lane holds column n0+(lane&15),
// contiguous K run of 16 at k0 + 16*(lane>=16).
static __device__ __forceinline__ v16h load_b_frag(const h16* Wt, int ldb, int n0, int k0, int lane) {
  const h16* p = Wt + (size_t)(n0 + (lane & 15)) * ldb + k0 + ((lane >> 4) << 4);
  v16h r;
#pragma unroll
  for (int i = 0; i < 16; ++i) r[i] = p[i];
  return r;
}

// B-fragment gathered from a row-major [K][N] LDS tile (rows are the K dim).
static __device__ __forceinline__ v16h load_b_rows(const h16* S, int lds_, int n0, int k0, int kvalid, int lane) {
  int n  = n0 + (lane & 15);
  int kb = k0 + ((lane >> 4) << 4);
  v16h r;
#pragma unroll
  for (int i = 0; i < 16; ++i) {
    int k = kb + i;
    r[i] = (k < kvalid) ? S[(size_t)k * lds_ + n] : (h16)0.0f;
  }
  return r;
}

// A-fragment = transpose of a row-major [K][M] LDS tile: A[m][k] = S[k][m0+m].
static __device__ __forceinline__ v16h load_a_tr(const h16* S, int lds_, int m0, int kvalid, int lane) {
  int m  = m0 + (lane & 15);
  int kb = (lane >> 4) << 4;
  v16h r;
#pragma unroll
  for (int i = 0; i < 8; ++i) {
    int k0 = kb + i, k1 = kb + 8 + i;
    r[i]     = (k0 < kvalid) ? S[(size_t)k0 * lds_ + m] : (h16)0.0f;
    r[8 + i] = (k1 < kvalid) ? S[(size_t)k1 * lds_ + m] : (h16)0.0f;
  }
  return r;
}

// C/D store: column n0+(lane&15), rows m0 + r + 8*(lane>=16).
static __device__ __forceinline__ void store_c_h16(h16* O, int ldo, int m0, int n0, v8f c, int lane) {
  int n  = n0 + (lane & 15);
  int mb = m0 + ((lane >> 4) << 3);
#pragma unroll
  for (int r = 0; r < 8; ++r) O[(size_t)(mb + r) * ldo + n] = (h16)c[r];
}

static __device__ __forceinline__ v8f mm_aw(const h16* A, int lda, int m0,
                                            const h16* Wt, int ldw, int n0,
                                            int K, int lane) {
  v8f c = vzero();
  for (int k = 0; k < K; k += 32)
    c = wmma32(load_a_frag(A, lda, m0, k, lane), load_b_frag(Wt, ldw, n0, k, lane), c);
  return c;
}

static __device__ __forceinline__ float red_max16(float v) {
#pragma unroll
  for (int m = 1; m < 16; m <<= 1) v = fmaxf(v, __shfl_xor(v, m, 32));
  return v;
}
static __device__ __forceinline__ float red_sum16(float v) {
#pragma unroll
  for (int m = 1; m < 16; m <<= 1) v += __shfl_xor(v, m, 32);
  return v;
}
static __device__ __forceinline__ float red_sum32(float v) {
#pragma unroll
  for (int m = 1; m < 32; m <<= 1) v += __shfl_xor(v, m, 32);
  return v;
}

static __device__ __forceinline__ float gelu_exact(float x) {
  return 0.5f * x * (1.0f + erff(x * 0.70710678118654752f));
}

// ---------------- K0: weight transpose + f32 -> f16 ----------------
__global__ void wprep_kernel(const float* __restrict__ src, h16* __restrict__ dst, int K, int N) {
  int i = blockIdx.x * blockDim.x + threadIdx.x;
  if (i < K * N) {
    int n = i / K, k = i % K;
    dst[i] = (h16)src[(size_t)k * N + n];   // dst[n][k] = src[k][n]
  }
}

// ---------------- K1: F-path, one wave per (b,t) ----------------
__global__ __launch_bounds__(32) void fpath_kernel(
    const float* __restrict__ x, const float* __restrict__ norm_w, const float* __restrict__ norm_b,
    const float* __restrict__ ln1w, const float* __restrict__ ln1b,
    const h16* __restrict__ WqT, const h16* __restrict__ WkT, const h16* __restrict__ WvT,
    const h16* __restrict__ WoT, const float* __restrict__ bo,
    const float* __restrict__ ln2w, const float* __restrict__ ln2b,
    const h16* __restrict__ W1T, const float* __restrict__ b1,
    const h16* __restrict__ W2T, const float* __restrict__ b2,
    const h16* __restrict__ WdT, const float* __restrict__ bd,
    float* __restrict__ wsY, float* __restrict__ wsStats)
{
  __shared__ float sRes[32][65];
  __shared__ float sX[32][65];
  __shared__ h16 sA[32][66];
  __shared__ h16 sS1[32][66];
  __shared__ h16 sS2[32][66];
  __shared__ h16 sCtx[64][66];

  const int lane = threadIdx.x;
  const int bt = blockIdx.x;
  const int b = bt / T_, t = bt % T_;

  // load tile: row f = lane, cols d; x layout [B,D,T,NF]
  const float* xb = x + (size_t)b * D_ * T_ * NF_ + (size_t)t * NF_ + lane;
  float s = 0.f, ss = 0.f;
  for (int d = 0; d < 64; ++d) {
    float v = xb[(size_t)d * T_ * NF_];
    sRes[lane][d] = v;
    s += v; ss += v * v;
  }
  s = red_sum32(s); ss = red_sum32(ss);
  float mu = s * (1.f / 2048.f);
  float rstd = rsqrtf(ss * (1.f / 2048.f) - mu * mu + EPS_);

  // ln2d -> x0 (sX), then per-row LN1 -> sA (lane owns full row f=lane)
  float rs = 0.f, rss = 0.f;
  for (int d = 0; d < 64; ++d) {
    float v = (sRes[lane][d] - mu) * rstd * norm_w[lane * 64 + d] + norm_b[lane * 64 + d];
    sX[lane][d] = v; rs += v; rss += v * v;
  }
  float rm = rs * (1.f / 64.f);
  float rr = rsqrtf(rss * (1.f / 64.f) - rm * rm + EPS_);
  for (int d = 0; d < 64; ++d)
    sA[lane][d] = (h16)(((sX[lane][d] - rm) * rr) * ln1w[d] + ln1b[d]);
  __syncthreads();

  // k (column softmax over n=32) -> sS1 ; v -> sS2
  for (int j = 0; j < 4; ++j) {
    v8f k0 = mm_aw(&sA[0][0], 66, 0,  WkT, 64, j * 16, 64, lane);
    v8f k1 = mm_aw(&sA[0][0], 66, 16, WkT, 64, j * 16, 64, lane);
    float cmax = -1e30f;
#pragma unroll
    for (int r = 0; r < 8; ++r) cmax = fmaxf(cmax, fmaxf(k0[r], k1[r]));
    cmax = fmaxf(cmax, __shfl_xor(cmax, 16, 32));
    float cs = 0.f;
#pragma unroll
    for (int r = 0; r < 8; ++r) {
      k0[r] = expf(k0[r] - cmax); k1[r] = expf(k1[r] - cmax);
      cs += k0[r] + k1[r];
    }
    cs += __shfl_xor(cs, 16, 32);
    float inv = 1.f / cs;
#pragma unroll
    for (int r = 0; r < 8; ++r) { k0[r] *= inv; k1[r] *= inv; }
    store_c_h16(&sS1[0][0], 66, 0,  j * 16, k0, lane);
    store_c_h16(&sS1[0][0], 66, 16, j * 16, k1, lane);
    v8f v0 = mm_aw(&sA[0][0], 66, 0,  WvT, 64, j * 16, 64, lane);
    v8f v1 = mm_aw(&sA[0][0], 66, 16, WvT, 64, j * 16, 64, lane);
    store_c_h16(&sS2[0][0], 66, 0,  j * 16, v0, lane);
    store_c_h16(&sS2[0][0], 66, 16, j * 16, v1, lane);
  }
  __syncthreads();

  // ctx (block-diagonal 64x64): ctx_h = k_soft^T @ v per head
  for (int i = lane; i < 64 * 66; i += 32) (&sCtx[0][0])[i] = (h16)0.0f;
  __syncthreads();
  for (int h = 0; h < 4; ++h) {
    v16h a = load_a_tr(&sS1[0][0], 66, h * 16, 32, lane);
    v16h bb = load_b_rows(&sS2[0][0], 66, h * 16, 0, 32, lane);
    v8f c = wmma32(a, bb, vzero());
    store_c_h16(&sCtx[0][0], 66, h * 16, h * 16, c, lane);
  }
  __syncthreads();

  // q with per-row (head) softmax -> sS1
  for (int j = 0; j < 4; ++j)
    for (int i = 0; i < 2; ++i) {
      v8f q = mm_aw(&sA[0][0], 66, i * 16, WqT, 64, j * 16, 64, lane);
#pragma unroll
      for (int r = 0; r < 8; ++r) {
        float mx = red_max16(q[r]);
        float e = expf(q[r] - mx);
        float sm = red_sum16(e);
        q[r] = e / sm * 0.25f;   // dh^-0.5, dh=16
      }
      store_c_h16(&sS1[0][0], 66, i * 16, j * 16, q, lane);
    }
  __syncthreads();

  // attnout = qs @ ctxBD -> sS2
  for (int j = 0; j < 4; ++j)
    for (int i = 0; i < 2; ++i) {
      v8f c = vzero();
      for (int ks = 0; ks < 2; ++ks)
        c = wmma32(load_a_frag(&sS1[0][0], 66, i * 16, ks * 32, lane),
                   load_b_rows(&sCtx[0][0], 66, j * 16, ks * 32, 64, lane), c);
      store_c_h16(&sS2[0][0], 66, i * 16, j * 16, c, lane);
    }
  __syncthreads();

  // x1 = x0 + attnout @ Wo + bo (accumulate into sX)
  for (int j = 0; j < 4; ++j)
    for (int i = 0; i < 2; ++i) {
      v8f c = mm_aw(&sS2[0][0], 66, i * 16, WoT, 64, j * 16, 64, lane);
      int n = j * 16 + (lane & 15);
      int mb = i * 16 + ((lane >> 4) << 3);
      float bv = bo[n];
#pragma unroll
      for (int r = 0; r < 8; ++r) sX[mb + r][n] += c[r] + bv;
    }
  __syncthreads();

  // LN2 rows -> sA
  {
    float as = 0.f, ass = 0.f;
    for (int d = 0; d < 64; ++d) { float v = sX[lane][d]; as += v; ass += v * v; }
    float am = as * (1.f / 64.f);
    float ar = rsqrtf(ass * (1.f / 64.f) - am * am + EPS_);
    for (int d = 0; d < 64; ++d)
      sA[lane][d] = (h16)(((sX[lane][d] - am) * ar) * ln2w[d] + ln2b[d]);
  }
  __syncthreads();

  // MLP: u accumulated over 4 hidden blocks of 64
  v8f accU[8];
#pragma unroll
  for (int q8 = 0; q8 < 8; ++q8) accU[q8] = vzero();
  for (int kb = 0; kb < 4; ++kb) {
    for (int jh = 0; jh < 4; ++jh)
      for (int i = 0; i < 2; ++i) {
        v8f hc = mm_aw(&sA[0][0], 66, i * 16, W1T, 64, kb * 64 + jh * 16, 64, lane);
        int n = jh * 16 + (lane & 15);
        float b1v = b1[kb * 64 + n];
#pragma unroll
        for (int r = 0; r < 8; ++r) hc[r] = gelu_exact(hc[r] + b1v);
        store_c_h16(&sS1[0][0], 66, i * 16, jh * 16, hc, lane);
      }
    __syncthreads();
    for (int j = 0; j < 4; ++j)
      for (int i = 0; i < 2; ++i)
        for (int ks = 0; ks < 2; ++ks)
          accU[j * 2 + i] = wmma32(load_a_frag(&sS1[0][0], 66, i * 16, ks * 32, lane),
                                   load_b_frag(W2T, 256, j * 16, kb * 64 + ks * 32, lane),
                                   accU[j * 2 + i]);
    __syncthreads();
  }
  for (int j = 0; j < 4; ++j)
    for (int i = 0; i < 2; ++i) {
      int n = j * 16 + (lane & 15);
      int mb = i * 16 + ((lane >> 4) << 3);
      float bv = b2[n];
#pragma unroll
      for (int r = 0; r < 8; ++r) sX[mb + r][n] += accU[j * 2 + i][r] + bv;
    }
  __syncthreads();

  // y = res + x2 @ Wd + bd
  for (int d = 0; d < 64; ++d) sA[lane][d] = (h16)sX[lane][d];
  __syncthreads();
  for (int j = 0; j < 4; ++j)
    for (int i = 0; i < 2; ++i) {
      v8f c = mm_aw(&sA[0][0], 66, i * 16, WdT, 64, j * 16, 64, lane);
      int n = j * 16 + (lane & 15);
      int mb = i * 16 + ((lane >> 4) << 3);
      float bv = bd[n];
#pragma unroll
      for (int r = 0; r < 8; ++r) sRes[mb + r][n] += c[r] + bv;
    }
  __syncthreads();

  // stats of y for the second ln2d + write y [B,T,F,D]
  float ys = 0.f, yss = 0.f;
  for (int d = 0; d < 64; ++d) { float v = sRes[lane][d]; ys += v; yss += v * v; }
  ys = red_sum32(ys); yss = red_sum32(yss);
  float ym = ys * (1.f / 2048.f);
  float yr = rsqrtf(yss * (1.f / 2048.f) - ym * ym + EPS_);
  if (lane == 0) { wsStats[bt * 2] = ym; wsStats[bt * 2 + 1] = yr; }
  float* yo = wsY + (size_t)bt * NF_ * D_;
  for (int fr = 0; fr < 32; ++fr) {
    yo[fr * 64 + lane]      = sRes[fr][lane];
    yo[fr * 64 + lane + 32] = sRes[fr][lane + 32];
  }
}

// ---------------- K2: T-path, 4 waves per (b,f) ----------------
#define KW 4
#define CH 16
__global__ __launch_bounds__(128) void tpath_kernel(
    const float* __restrict__ wsY, const float* __restrict__ wsStats,
    const float* __restrict__ norm_w, const float* __restrict__ norm_b,
    const float* __restrict__ ln1w, const float* __restrict__ ln1b,
    const h16* __restrict__ WqT, const h16* __restrict__ WkT, const h16* __restrict__ WvT,
    const h16* __restrict__ WoT, const float* __restrict__ bo,
    const float* __restrict__ ln2w, const float* __restrict__ ln2b,
    const h16* __restrict__ W1T, const float* __restrict__ b1,
    const h16* __restrict__ W2T, const float* __restrict__ b2,
    const h16* __restrict__ WdT, const float* __restrict__ bd,
    float* __restrict__ out)
{
  __shared__ float sXr[KW][CH][65];
  __shared__ h16 sAa[KW][CH][66];
  __shared__ h16 sB1[KW][CH][66];
  __shared__ h16 sB2[KW][CH][66];
  __shared__ h16 sCtxBD[64][66];
  __shared__ float sPart[KW][64];
  __shared__ float sColMax[64];
  __shared__ float sColSum[64];
  __shared__ float sCtx[4][16][16];

  const int tid = threadIdx.x;
  const int w = tid >> 5, lane = tid & 31;
  const int bf = blockIdx.x;
  const int b = bf / NF_, f = bf % NF_;
  const int NCH = T_ / CH;   // 125, exact

  auto build_a = [&](int c, bool need_x) {
    for (int r = 0; r < CH; ++r) {
      int t = c * CH + r;
      const float* yr = wsY + ((size_t)(b * T_ + t) * NF_ + f) * D_;
      float mu = wsStats[(b * T_ + t) * 2];
      float rstd = wsStats[(b * T_ + t) * 2 + 1];
      float v0 = yr[lane], v1 = yr[lane + 32];
      float x0a = (v0 - mu) * rstd * norm_w[f * 64 + lane] + norm_b[f * 64 + lane];
      float x0b = (v1 - mu) * rstd * norm_w[f * 64 + lane + 32] + norm_b[f * 64 + lane + 32];
      if (need_x) { sXr[w][r][lane] = x0a; sXr[w][r][lane + 32] = x0b; }
      float rm = red_sum32(x0a + x0b) * (1.f / 64.f);
      float d0 = x0a - rm, d1 = x0b - rm;
      float rr = rsqrtf(red_sum32(d0 * d0 + d1 * d1) * (1.f / 64.f) + EPS_);
      sAa[w][r][lane]      = (h16)(d0 * rr * ln1w[lane] + ln1b[lane]);
      sAa[w][r][lane + 32] = (h16)(d1 * rr * ln1w[lane + 32] + ln1b[lane + 32]);
    }
  };

  // ---- Phase 1: global k-column max over T ----
  float cmax[4] = {-1e30f, -1e30f, -1e30f, -1e30f};
  for (int c = w; c < NCH; c += KW) {
    build_a(c, false);
    for (int j = 0; j < 4; ++j) {
      v8f k = mm_aw(&sAa[w][0][0], 66, 0, WkT, 64, j * 16, 64, lane);
#pragma unroll
      for (int r = 0; r < 8; ++r) cmax[j] = fmaxf(cmax[j], k[r]);
    }
  }
  for (int j = 0; j < 4; ++j) {
    float v = fmaxf(cmax[j], __shfl_xor(cmax[j], 16, 32));
    if (lane < 16) sPart[w][j * 16 + lane] = v;
  }
  for (int i = tid; i < 4 * 256; i += 128) (&sCtx[0][0][0])[i] = 0.f;
  __syncthreads();
  if (tid < 64) {
    float m = sPart[0][tid];
    for (int ww = 1; ww < KW; ++ww) m = fmaxf(m, sPart[ww][tid]);
    sColMax[tid] = m;
  }
  __syncthreads();

  // ---- Phase 2: ctx = sum_n exp(k - max) (x) v ; column sums ----
  float csum[4] = {0.f, 0.f, 0.f, 0.f};
  v8f ctxA[4];
#pragma unroll
  for (int h = 0; h < 4; ++h) ctxA[h] = vzero();
  for (int c = w; c < NCH; c += KW) {
    build_a(c, false);
    for (int j = 0; j < 4; ++j) {
      v8f k = mm_aw(&sAa[w][0][0], 66, 0, WkT, 64, j * 16, 64, lane);
      int n = j * 16 + (lane & 15);
      float cm = sColMax[n];
#pragma unroll
      for (int r = 0; r < 8; ++r) { float e = expf(k[r] - cm); k[r] = e; csum[j] += e; }
      store_c_h16(&sB1[w][0][0], 66, 0, j * 16, k, lane);
      v8f vv = mm_aw(&sAa[w][0][0], 66, 0, WvT, 64, j * 16, 64, lane);
      store_c_h16(&sB2[w][0][0], 66, 0, j * 16, vv, lane);
    }
    for (int h = 0; h < 4; ++h)
      ctxA[h] = wmma32(load_a_tr(&sB1[w][0][0], 66, h * 16, CH, lane),
                       load_b_rows(&sB2[w][0][0], 66, h * 16, 0, CH, lane), ctxA[h]);
  }
  for (int j = 0; j < 4; ++j) {
    float v = csum[j] + __shfl_xor(csum[j], 16, 32);
    if (lane < 16) sPart[w][j * 16 + lane] = v;
  }
  for (int h = 0; h < 4; ++h) {
    int n = lane & 15;
    int mb = (lane >> 4) << 3;
#pragma unroll
    for (int r = 0; r < 8; ++r) atomicAdd(&sCtx[h][mb + r][n], ctxA[h][r]);
  }
  __syncthreads();
  if (tid < 64) {
    float sum = 0.f;
    for (int ww = 0; ww < KW; ++ww) sum += sPart[ww][tid];
    sColSum[tid] = sum;
  }
  __syncthreads();
  for (int i = tid; i < 4096; i += 128) {
    int d = i >> 6, e = i & 63;
    sCtxBD[d][e] = ((d >> 4) == (e >> 4))
                     ? (h16)(sCtx[d >> 4][d & 15][e & 15] / sColSum[d])
                     : (h16)0.0f;
  }
  __syncthreads();

  // ---- Phase 3: per-chunk outputs ----
  for (int c = w; c < NCH; c += KW) {
    build_a(c, true);
    // q softmax
    for (int j = 0; j < 4; ++j) {
      v8f q = mm_aw(&sAa[w][0][0], 66, 0, WqT, 64, j * 16, 64, lane);
#pragma unroll
      for (int r = 0; r < 8; ++r) {
        float mx = red_max16(q[r]);
        float e = expf(q[r] - mx);
        float sm = red_sum16(e);
        q[r] = e / sm * 0.25f;
      }
      store_c_h16(&sB1[w][0][0], 66, 0, j * 16, q, lane);
    }
    // attnout = qs @ ctxBD
    for (int j = 0; j < 4; ++j) {
      v8f ao = vzero();
      for (int ks = 0; ks < 2; ++ks)
        ao = wmma32(load_a_frag(&sB1[w][0][0], 66, 0, ks * 32, lane),
                    load_b_rows(&sCtxBD[0][0], 66, j * 16, ks * 32, 64, lane), ao);
      store_c_h16(&sB2[w][0][0], 66, 0, j * 16, ao, lane);
    }
    // x1 = x0 + attnout @ Wo + bo
    for (int j = 0; j < 4; ++j) {
      v8f o = mm_aw(&sB2[w][0][0], 66, 0, WoT, 64, j * 16, 64, lane);
      int n = j * 16 + (lane & 15);
      int mb = (lane >> 4) << 3;
      float bv = bo[n];
#pragma unroll
      for (int r = 0; r < 8; ++r) sXr[w][mb + r][n] += o[r] + bv;
    }
    // LN2 rows -> sAa
    for (int r = 0; r < CH; ++r) {
      float a0 = sXr[w][r][lane], a1 = sXr[w][r][lane + 32];
      float m = red_sum32(a0 + a1) * (1.f / 64.f);
      float d0 = a0 - m, d1 = a1 - m;
      float rr = rsqrtf(red_sum32(d0 * d0 + d1 * d1) * (1.f / 64.f) + EPS_);
      sAa[w][r][lane]      = (h16)(d0 * rr * ln2w[lane] + ln2b[lane]);
      sAa[w][r][lane + 32] = (h16)(d1 * rr * ln2w[lane + 32] + ln2b[lane + 32]);
    }
    // MLP fused over hidden blocks
    v8f accU[4];
#pragma unroll
    for (int j = 0; j < 4; ++j) accU[j] = vzero();
    for (int kb = 0; kb < 4; ++kb) {
      for (int jh = 0; jh < 4; ++jh) {
        v8f hc = mm_aw(&sAa[w][0][0], 66, 0, W1T, 64, kb * 64 + jh * 16, 64, lane);
        int n = jh * 16 + (lane & 15);
        float b1v = b1[kb * 64 + n];
#pragma unroll
        for (int r = 0; r < 8; ++r) hc[r] = gelu_exact(hc[r] + b1v);
        store_c_h16(&sB1[w][0][0], 66, 0, jh * 16, hc, lane);
      }
      for (int j = 0; j < 4; ++j)
        for (int ks = 0; ks < 2; ++ks)
          accU[j] = wmma32(load_a_frag(&sB1[w][0][0], 66, 0, ks * 32, lane),
                           load_b_frag(W2T, 256, j * 16, kb * 64 + ks * 32, lane), accU[j]);
    }
    for (int j = 0; j < 4; ++j) {
      int n = j * 16 + (lane & 15);
      int mb = (lane >> 4) << 3;
      float bv = b2[n];
#pragma unroll
      for (int r = 0; r < 8; ++r) sXr[w][mb + r][n] += accU[j][r] + bv;
    }
    // x2 rows -> f16
    for (int r = 0; r < CH; ++r) {
      sAa[w][r][lane]      = (h16)sXr[w][r][lane];
      sAa[w][r][lane + 32] = (h16)sXr[w][r][lane + 32];
    }
    // final = y + x2 @ Wd + bd ; write out [B,D,T,NF]
    for (int j = 0; j < 4; ++j) {
      v8f oo = mm_aw(&sAa[w][0][0], 66, 0, WdT, 64, j * 16, 64, lane);
      int n = j * 16 + (lane & 15);
      int mb = (lane >> 4) << 3;
      float bv = bd[n];
#pragma unroll
      for (int r = 0; r < 8; ++r) {
        int t = c * CH + mb + r;
        float y = wsY[((size_t)(b * T_ + t) * NF_ + f) * D_ + n];
        out[(((size_t)b * D_ + n) * T_ + t) * NF_ + f] = y + oo[r] + bv;
      }
    }
  }
}

extern "C" void kernel_launch(void* const* d_in, const int* in_sizes, int n_in,
                              void* d_out, int out_size, void* d_ws, size_t ws_size,
                              hipStream_t stream) {
  (void)in_sizes; (void)n_in; (void)out_size; (void)ws_size;
  const float* x      = (const float*)d_in[0];
  const float* norm_w = (const float*)d_in[1];
  const float* norm_b = (const float*)d_in[2];

  float* wsY = (float*)d_ws;
  size_t NY = (size_t)B_ * T_ * NF_ * D_;
  float* wsStats = wsY + NY;
  h16* wpool = (h16*)(wsStats + (size_t)B_ * T_ * 2);
  h16* fW = wpool;
  h16* tW = wpool + 53248;

  struct Job { int idx; int K; int N; int off; };
  const Job jobs[7] = {
    {5, 64, 64, 0},        // Wq
    {6, 64, 64, 4096},     // Wk
    {7, 64, 64, 8192},     // Wv
    {8, 64, 64, 12288},    // Wo
    {12, 64, 256, 16384},  // W1  -> W1T [256][64]
    {14, 256, 64, 32768},  // W2  -> W2T [64][256]
    {16, 64, 64, 49152}    // Wd
  };
  for (int p = 0; p < 2; ++p) {
    int base = p * 15;
    h16* wp = p ? tW : fW;
    for (int m = 0; m < 7; ++m) {
      int n = jobs[m].K * jobs[m].N;
      wprep_kernel<<<(n + 255) / 256, 256, 0, stream>>>(
          (const float*)d_in[jobs[m].idx + base], wp + jobs[m].off, jobs[m].K, jobs[m].N);
    }
  }

  fpath_kernel<<<B_ * T_, 32, 0, stream>>>(
      x, norm_w, norm_b,
      (const float*)d_in[3], (const float*)d_in[4],
      fW + 0, fW + 4096, fW + 8192, fW + 12288, (const float*)d_in[9],
      (const float*)d_in[10], (const float*)d_in[11],
      fW + 16384, (const float*)d_in[13],
      fW + 32768, (const float*)d_in[15],
      fW + 49152, (const float*)d_in[17],
      wsY, wsStats);

  tpath_kernel<<<B_ * NF_, 128, 0, stream>>>(
      wsY, wsStats, norm_w, norm_b,
      (const float*)d_in[18], (const float*)d_in[19],
      tW + 0, tW + 4096, tW + 8192, tW + 12288, (const float*)d_in[24],
      (const float*)d_in[25], (const float*)d_in[26],
      tW + 16384, (const float*)d_in[28],
      tW + 32768, (const float*)d_in[30],
      tW + 49152, (const float*)d_in[32],
      (float*)d_out);
}
